// UnSwagAttentionLayer_70042326663827
// MI455X (gfx1250) — matrix operand order repaired and checked by reading, not compiled, not run on hardware
//
#include <hip/hip_runtime.h>
#include <cstdint>
#include <cstddef>

#define B_  2
#define S_  2048
#define D_  768
#define H_  12
#define HD_ 64
#define FF_ 3072
#define KS_ 5

typedef __bf16 bf16;
typedef bf16  v16bf __attribute__((ext_vector_type(16)));
typedef float v8f   __attribute__((ext_vector_type(8)));
typedef unsigned int v4u __attribute__((ext_vector_type(4)));
typedef int v4i __attribute__((ext_vector_type(4)));
typedef int v8i __attribute__((ext_vector_type(8)));

#if defined(__has_builtin)
#if __has_builtin(__builtin_amdgcn_tensor_load_to_lds)
#define HAVE_TDM 1
#endif
#endif
#ifndef HAVE_TDM
#define HAVE_TDM 0
#endif

static __device__ __forceinline__ v8f wmma_bf16(v16bf a, v16bf b, v8f c) {
  return __builtin_amdgcn_wmma_f32_16x16x32_bf16(false, a, false, b, (short)0, c,
                                                 false, false);
}

static __device__ __forceinline__ float gelu_exact(float x) {
  return 0.5f * x * (1.0f + erff(x * 0.70710678118654752f));
}

static __device__ __forceinline__ int akmap(int e, int half) {
  return ((e < 8) ? e : (e + 8)) + half * 8;
}

// LDS byte offset of a __shared__ object (workgroup-relative)
template <typename T>
static __device__ __forceinline__ unsigned lds_off(T* p) {
  return (unsigned)(unsigned long long)(__attribute__((address_space(3))) T*)p;
}

// Issue one TDM 2-D tile load: bf16 elements, tile == tensor slice (no OOB),
// row pitch strideElems.  D# packed per CDNA5 ISA.
static __device__ __forceinline__ void tdm_load_tile_bf16(
    const bf16* gsrc, unsigned ldsOffBytes, int tileW, int tileH,
    long long strideElems) {
#if HAVE_TDM
  unsigned long long ga = (unsigned long long)(const void*)gsrc;
  v4u g0;
  g0[0] = 1u;                                   // count=1, user descriptor
  g0[1] = ldsOffBytes;                          // lds_addr
  g0[2] = (unsigned)ga;                         // global_addr[31:0]
  g0[3] = (unsigned)((ga >> 32) & 0x01FFFFFFu)  // global_addr[56:32]
          | (2u << 30);                         // type = 2 ("image")
  v8i g1;
  g1[0] = 1 << 16;                                   // data_size=1 (2 bytes)
  g1[1] = (tileW & 0xFFFF) << 16;                    // tensor_dim0 lo
  g1[2] = ((unsigned)tileW >> 16) | ((tileH & 0xFFFF) << 16);  // td0 hi|td1 lo
  g1[3] = ((unsigned)tileH >> 16) | ((tileW & 0xFFFF) << 16);  // td1 hi|tile_dim0
  g1[4] = tileH & 0xFFFF;                            // tile_dim1 (tile_dim2=0)
  g1[5] = (int)(strideElems & 0xFFFFFFFFll);         // tensor_dim0_stride lo
  g1[6] = (int)((strideElems >> 32) & 0xFFFF);       // stride hi | td1_stride=0
  g1[7] = 0;
  v4i gz = (v4i)0;
#if __clang_major__ >= 23
  __builtin_amdgcn_tensor_load_to_lds(g0, g1, gz, gz, (v8i)0, 0);
#else
  __builtin_amdgcn_tensor_load_to_lds(g0, g1, gz, gz, 0);
#endif
#else
  (void)gsrc; (void)ldsOffBytes; (void)tileW; (void)tileH; (void)strideElems;
#endif
}

static __device__ __forceinline__ void tdm_wait() {
#if HAVE_TDM
  __builtin_amdgcn_s_wait_tensorcnt(0);
#endif
}

// ---------------------------------------------------------------------------
// 0) zero counters
// ---------------------------------------------------------------------------
__global__ void init_kernel(int* counters) {
  if (threadIdx.x < 8) counters[threadIdx.x] = 0;
}

// ---------------------------------------------------------------------------
// 0b) f32 -> bf16 conversion (weights / activations that feed WMMA)
// ---------------------------------------------------------------------------
__global__ __launch_bounds__(256) void cvt_kernel(const float* __restrict__ in,
                                                  bf16* __restrict__ out,
                                                  long n) {
  long i = (long)blockIdx.x * 256 + threadIdx.x;
  if (i < n) out[i] = (bf16)in[i];
}

// ---------------------------------------------------------------------------
// 1) Router MLP 768->64->4, softmax/argmax, route codes + n10/n11 counters
// ---------------------------------------------------------------------------
__global__ __launch_bounds__(64) void router_kernel(
    const float* __restrict__ h, const float* __restrict__ w1,
    const float* __restrict__ b1, const float* __restrict__ w2,
    const float* __restrict__ b2, int* __restrict__ codes,
    int* __restrict__ counters) {
  __shared__ float hid[64];
  __shared__ float lg[4];
  int tok = blockIdx.x;
  int j = threadIdx.x;
  const float* hr = h + (size_t)tok * D_;
  float acc = b1[j];
  for (int d = 0; d < D_; ++d) acc = fmaf(hr[d], w1[d * 64 + j], acc);
  hid[j] = gelu_exact(acc);
  __syncthreads();
  if (j < 4) {
    float a = b2[j];
    for (int i = 0; i < 64; ++i) a = fmaf(hid[i], w2[i * 4 + j], a);
    lg[j] = a;
  }
  __syncthreads();
  if (j == 0) {
    float mx = fmaxf(fmaxf(lg[0], lg[1]), fmaxf(lg[2], lg[3]));
    float e0 = __expf(lg[0] - mx), e1 = __expf(lg[1] - mx);
    float e2 = __expf(lg[2] - mx), e3 = __expf(lg[3] - mx);
    float sum = e0 + e1 + e2 + e3;
    int am = 0;
    float bmax = lg[0];
    if (lg[1] > bmax) { bmax = lg[1]; am = 1; }
    if (lg[2] > bmax) { bmax = lg[2]; am = 2; }
    if (lg[3] > bmax) { bmax = lg[3]; am = 3; }
    int code = am;
    if (am == 0 && (e0 / sum) > 0.99f) code |= 4;
    codes[tok] = code;
    int b = tok / S_;
    if (am == 2) atomicAdd(&counters[b], 1);
    if (am == 3) atomicAdd(&counters[2 + b], 1);
  }
}

// ---------------------------------------------------------------------------
// 2) depthwise conv (KS=5, same pad) + GELU -> bf16
// ---------------------------------------------------------------------------
__global__ __launch_bounds__(256) void dwconv_kernel(
    const float* __restrict__ h, const float* __restrict__ cw,
    const float* __restrict__ cb, bf16* __restrict__ out) {
  size_t idx = (size_t)blockIdx.x * 256 + threadIdx.x;
  int d = (int)(idx % D_);
  int s = (int)((idx / D_) % S_);
  int b = (int)(idx / ((size_t)D_ * S_));
  float acc = cb[d];
#pragma unroll
  for (int t = 0; t < KS_; ++t) {
    int si = s + t - KS_ / 2;
    if (si >= 0 && si < S_)
      acc = fmaf(h[((size_t)(b * S_ + si)) * D_ + d], cw[d * KS_ + t], acc);
  }
  out[idx] = (bf16)gelu_exact(acc);
}

// ---------------------------------------------------------------------------
// 3) WMMA GEMM, bf16 operands, TDM-staged LDS tiles.
//    C = act(A[M,K] @ W[K,N] + bias); f32 and/or bf16 outputs (nullable).
// ---------------------------------------------------------------------------
#define GBM 128
#define GBN 64
#define GBK 32
__global__ __launch_bounds__(256) void gemm_kernel(
    const bf16* __restrict__ A, const bf16* __restrict__ W,
    const float* __restrict__ bias, float* __restrict__ Cf,
    bf16* __restrict__ Cb, int M, int N, int K, int act) {
  __shared__ bf16 As[GBM][GBK];
  __shared__ bf16 Bs[GBK][GBN];
  int tid = threadIdx.x;
  int lane = tid & 31, wid = tid >> 5;
  int wm = wid >> 1, wn = wid & 1;
  int bm = blockIdx.x * GBM, bn = blockIdx.y * GBN;
  int half = lane >> 4, l16 = lane & 15;

  v8f acc[2][2];
#pragma unroll
  for (int i = 0; i < 2; ++i)
#pragma unroll
    for (int j = 0; j < 2; ++j) acc[i][j] = (v8f)0.0f;

  for (int k0 = 0; k0 < K; k0 += GBK) {
    __syncthreads();  // previous iteration's consumers done
#if HAVE_TDM
    if (tid < 32) {  // wave 0 drives the Tensor Data Mover
      tdm_load_tile_bf16(A + (size_t)bm * K + k0, lds_off(&As[0][0]), GBK, GBM,
                         K);
      tdm_load_tile_bf16(W + (size_t)k0 * N + bn, lds_off(&Bs[0][0]), GBN, GBK,
                         N);
      tdm_wait();
    }
#else
#pragma unroll
    for (int i = 0; i < 16; ++i) {
      int idx = tid + i * 256;
      int r = idx >> 5, c = idx & 31;
      As[r][c] = A[(size_t)(bm + r) * K + k0 + c];
    }
#pragma unroll
    for (int i = 0; i < 8; ++i) {
      int idx = tid + i * 256;
      int r = idx >> 6, c = idx & 63;
      Bs[r][c] = W[(size_t)(k0 + r) * N + bn + c];
    }
#endif
    __syncthreads();

    v16bf af[2], bb[2];
#pragma unroll
    for (int i = 0; i < 2; ++i) {
      int m0 = wm * 32 + i * 16;
#pragma unroll
      for (int e = 0; e < 16; ++e) af[i][e] = As[m0 + l16][akmap(e, half)];
    }
#pragma unroll
    for (int j = 0; j < 2; ++j) {
      int n0 = wn * 32 + j * 16;
#pragma unroll
      for (int e = 0; e < 16; ++e) bb[j][e] = Bs[akmap(e, half)][n0 + l16];
    }
#pragma unroll
    for (int i = 0; i < 2; ++i)
#pragma unroll
      for (int j = 0; j < 2; ++j) acc[i][j] = wmma_bf16(af[i], bb[j], acc[i][j]);
  }

#pragma unroll
  for (int i = 0; i < 2; ++i)
#pragma unroll
    for (int j = 0; j < 2; ++j)
#pragma unroll
      for (int r = 0; r < 8; ++r) {
        int row = bm + wm * 32 + i * 16 + r + half * 8;
        int col = bn + wn * 32 + j * 16 + l16;
        float v = acc[i][j][r] + bias[col];
        if (act == 1) v = gelu_exact(v);
        if (Cf) Cf[(size_t)row * N + col] = v;
        if (Cb) Cb[(size_t)row * N + col] = (bf16)v;
      }
}

// ---------------------------------------------------------------------------
// 4) fused (x [+res]) LayerNorm over D=768; optional extra bf16 output
// ---------------------------------------------------------------------------
__global__ __launch_bounds__(256) void rowln_kernel(
    const float* __restrict__ x, const float* __restrict__ res,
    const float* __restrict__ g, const float* __restrict__ bta,
    float* __restrict__ out, bf16* __restrict__ outb, int D) {
  __shared__ float red[256];
  int row = blockIdx.x, tid = threadIdx.x;
  float v[3];
  float s = 0.f;
#pragma unroll
  for (int i = 0; i < 3; ++i) {
    int d = tid + i * 256;
    float t = x[(size_t)row * D + d];
    if (res) t += res[(size_t)row * D + d];
    v[i] = t;
    s += t;
  }
  red[tid] = s;
  __syncthreads();
  for (int off = 128; off > 0; off >>= 1) {
    if (tid < off) red[tid] += red[tid + off];
    __syncthreads();
  }
  float mean = red[0] / (float)D;
  __syncthreads();
  s = 0.f;
#pragma unroll
  for (int i = 0; i < 3; ++i) {
    float t = v[i] - mean;
    s += t * t;
  }
  red[tid] = s;
  __syncthreads();
  for (int off = 128; off > 0; off >>= 1) {
    if (tid < off) red[tid] += red[tid + off];
    __syncthreads();
  }
  float inv = rsqrtf(red[0] / (float)D + 1e-5f);
  __syncthreads();
#pragma unroll
  for (int i = 0; i < 3; ++i) {
    int d = tid + i * 256;
    float o = (v[i] - mean) * inv * g[d] + bta[d];
    if (out) out[(size_t)row * D + d] = o;
    if (outb) outb[(size_t)row * D + d] = (bf16)o;
  }
}

// ---------------------------------------------------------------------------
// 5) register update: amean over m10 tokens -> regpre = reg + a*(amean-reg)
// ---------------------------------------------------------------------------
__global__ __launch_bounds__(256) void regupd_kernel(
    const float* __restrict__ proj, const int* __restrict__ codes,
    const int* __restrict__ counters, const float* __restrict__ reg,
    const float* __restrict__ alpha, float* __restrict__ regpre) {
  __shared__ unsigned char flags[S_];
  int b = blockIdx.y;
  int d = blockIdx.x * 256 + threadIdx.x;
  for (int i = 0; i < S_ / 256; ++i) {
    int s = threadIdx.x + i * 256;
    flags[s] = (unsigned char)((codes[b * S_ + s] & 3) == 2);
  }
  __syncthreads();
  float sum = 0.f;
  for (int s = 0; s < S_; ++s)
    if (flags[s]) sum += proj[((size_t)(b * S_ + s)) * D_ + d];
  int cnt = counters[b];
  if (cnt < 1) cnt = 1;
  float amean = sum / (float)cnt;
  float a = 1.0f / (1.0f + __expf(-alpha[0]));
  float r0 = reg[b * D_ + d];
  regpre[b * D_ + d] = r0 + a * (amean - r0);
}

// ---------------------------------------------------------------------------
// 6) Kr/Vr projections of register_state (tiny GEMM, VALU)
// ---------------------------------------------------------------------------
__global__ __launch_bounds__(256) void regkv_kernel(
    const float* __restrict__ reg, const float* __restrict__ rkw,
    const float* __restrict__ rkb, const float* __restrict__ rvw,
    const float* __restrict__ rvb, float* __restrict__ Kr,
    float* __restrict__ Vr) {
  int idx = blockIdx.x * 256 + threadIdx.x;
  int which = idx / (B_ * D_);
  int rem = idx % (B_ * D_);
  int b = rem / D_, d = rem % D_;
  const float* w = which ? rvw : rkw;
  float acc = which ? rvb[d] : rkb[d];
  const float* rr = reg + (size_t)b * D_;
  for (int k = 0; k < D_; ++k) acc = fmaf(rr[k], w[(size_t)k * D_ + d], acc);
  (which ? Vr : Kr)[b * D_ + d] = acc;
}

// ---------------------------------------------------------------------------
// 7) flash attention over m11 keys + pad column + register column
//    grid (S/64, H, B), 128 threads = 4 waves; K/V chunks staged by TDM
// ---------------------------------------------------------------------------
__global__ __launch_bounds__(128) void attn_kernel(
    const bf16* __restrict__ Qg, const bf16* __restrict__ Kg,
    const bf16* __restrict__ Vg, const float* __restrict__ kb,
    const float* __restrict__ vb, const float* __restrict__ Kr,
    const float* __restrict__ Vr, const int* __restrict__ codes,
    const int* __restrict__ counters, bf16* __restrict__ out) {
  const float scale = 0.125f;
  int b = blockIdx.z, h = blockIdx.y;
  int lane = threadIdx.x & 31, wid = threadIdx.x >> 5;
  int half = lane >> 4, l16 = lane & 15;
  int q0 = blockIdx.x * 64 + wid * 16;

  __shared__ bf16 Kc[32][64];
  __shared__ bf16 Vc[32][64];
  __shared__ int fl[32];
  __shared__ bf16 Pb[4][16][32];
  __shared__ float bc[4][4][16];

  v16bf qa0, qa1;
  {
    const bf16* Qrow = Qg + ((size_t)(b * S_ + q0 + l16)) * D_ + h * HD_;
#pragma unroll
    for (int e = 0; e < 16; ++e) {
      int f = akmap(e, half);
      qa0[e] = Qrow[f];
      qa1[e] = Qrow[f + 32];
    }
  }

  float m8[8], l8[8];
  v8f acc[4];
#pragma unroll
  for (int r = 0; r < 8; ++r) { m8[r] = -1e30f; l8[r] = 0.f; }
#pragma unroll
  for (int nt = 0; nt < 4; ++nt) acc[nt] = (v8f)0.0f;

  for (int kc = 0; kc < S_; kc += 32) {
    __syncthreads();
#if HAVE_TDM
    if (threadIdx.x < 32) {
      size_t g = ((size_t)(b * S_ + kc)) * D_ + h * HD_;
      tdm_load_tile_bf16(Kg + g, lds_off(&Kc[0][0]), HD_, 32, D_);
      tdm_load_tile_bf16(Vg + g, lds_off(&Vc[0][0]), HD_, 32, D_);
      tdm_wait();
    }
#else
#pragma unroll
    for (int i = 0; i < 16; ++i) {
      int idx = threadIdx.x + i * 128;
      int r = idx >> 6, c = idx & 63;
      size_t g = ((size_t)(b * S_ + kc + r)) * D_ + h * HD_ + c;
      Kc[r][c] = Kg[g];
      Vc[r][c] = Vg[g];
    }
#endif
    if (threadIdx.x < 32) fl[threadIdx.x] = codes[b * S_ + kc + threadIdx.x];
    __syncthreads();

    v8f s0 = (v8f)0.0f, s1 = (v8f)0.0f;
    {
      v16bf bk0a, bk0b, bk1a, bk1b;
#pragma unroll
      for (int e = 0; e < 16; ++e) {
        bk0a[e] = Kc[l16][akmap(e, half)];
        bk0b[e] = Kc[l16][32 + akmap(e, half)];
        bk1a[e] = Kc[16 + l16][akmap(e, half)];
        bk1b[e] = Kc[16 + l16][32 + akmap(e, half)];
      }
      s0 = wmma_bf16(qa0, bk0a, s0);
      s0 = wmma_bf16(qa1, bk0b, s0);
      s1 = wmma_bf16(qa0, bk1a, s1);
      s1 = wmma_bf16(qa1, bk1b, s1);
    }
    bool valid0 = (fl[l16] & 3) == 3;
    bool valid1 = (fl[16 + l16] & 3) == 3;

#pragma unroll
    for (int r = 0; r < 8; ++r) {
      float a0 = valid0 ? s0[r] * scale : -1e30f;
      float a1 = valid1 ? s1[r] * scale : -1e30f;
      float t = fmaxf(a0, a1);
      t = fmaxf(t, __shfl_xor(t, 1, 32));
      t = fmaxf(t, __shfl_xor(t, 2, 32));
      t = fmaxf(t, __shfl_xor(t, 4, 32));
      t = fmaxf(t, __shfl_xor(t, 8, 32));
      float mn = fmaxf(m8[r], t);
      float fs = __expf(m8[r] - mn);
      m8[r] = mn;
      float p0 = (a0 < -1e29f) ? 0.f : __expf(a0 - mn);
      float p1 = (a1 < -1e29f) ? 0.f : __expf(a1 - mn);
      float rs = p0 + p1;
      rs += __shfl_xor(rs, 1, 32);
      rs += __shfl_xor(rs, 2, 32);
      rs += __shfl_xor(rs, 4, 32);
      rs += __shfl_xor(rs, 8, 32);
      l8[r] = l8[r] * fs + rs;
#pragma unroll
      for (int nt = 0; nt < 4; ++nt) acc[nt][r] *= fs;
      int row = r + half * 8;
      Pb[wid][row][l16] = (bf16)p0;
      Pb[wid][row][16 + l16] = (bf16)p1;
    }
    v16bf pa;
#pragma unroll
    for (int e = 0; e < 16; ++e) pa[e] = Pb[wid][l16][akmap(e, half)];
#pragma unroll
    for (int nt = 0; nt < 4; ++nt) {
      v16bf bv;
#pragma unroll
      for (int e = 0; e < 16; ++e) bv[e] = Vc[akmap(e, half)][nt * 16 + l16];
      acc[nt] = wmma_bf16(pa, bv, acc[nt]);
    }
  }

  // ---- pad-bias column + register column ----
  int n11_0 = counters[2], n11_1 = counters[3];
  int nmax = n11_0 > n11_1 ? n11_0 : n11_1;
  int padc = nmax - (b == 0 ? n11_0 : n11_1);
  float logpad = padc > 0 ? __logf((float)padc) : -1e30f;
  const float* K0 = kb + h * HD_;
  const float* V0 = vb + h * HD_;
  const float* Krh = Kr + (size_t)b * D_ + h * HD_;
  const float* Vrh = Vr + (size_t)b * D_ + h * HD_;
  float dp = 0.f, dr = 0.f;
#pragma unroll
  for (int e = 0; e < 16; ++e) {
    int f = akmap(e, half);
    dp += (float)qa0[e] * K0[f] + (float)qa1[e] * K0[f + 32];
    dr += (float)qa0[e] * Krh[f] + (float)qa1[e] * Krh[f + 32];
  }
  dp += __shfl_xor(dp, 16, 32);
  dr += __shfl_xor(dr, 16, 32);
  float s_pad = dp * scale + logpad;
  float s_reg = dr * scale;

  if (l16 == 0) {
#pragma unroll
    for (int r = 0; r < 8; ++r) {
      bc[wid][0][r + half * 8] = m8[r];
      bc[wid][1][r + half * 8] = l8[r];
    }
  }
  float mrow = bc[wid][0][l16];
  float lrow = bc[wid][1][l16];
  float mn = fmaxf(mrow, fmaxf(s_pad, s_reg));
  float so = __expf(mrow - mn);
  float wp = (s_pad < -1e29f) ? 0.f : __expf(s_pad - mn);
  float wr = __expf(s_reg - mn);
  float lt = lrow * so + wp + wr;
  if (lane < 16) {
    bc[wid][0][lane] = so;
    bc[wid][1][lane] = wp;
    bc[wid][2][lane] = wr;
    bc[wid][3][lane] = lt;
  }
#pragma unroll
  for (int r = 0; r < 8; ++r) {
    int row = r + half * 8;
    float so_r = bc[wid][0][row];
    float wp_r = bc[wid][1][row];
    float wr_r = bc[wid][2][row];
    float lt_r = bc[wid][3][row];
#pragma unroll
    for (int nt = 0; nt < 4; ++nt) {
      int col = nt * 16 + l16;
      float v = (acc[nt][r] * so_r + wp_r * V0[col] + wr_r * Vrh[col]) / lt_r;
      out[((size_t)(b * S_ + q0 + row)) * D_ + h * HD_ + col] = (bf16)v;
    }
  }
}

// ---------------------------------------------------------------------------
// 8) combine pathways + residual:  ypre = select(code) + h
// ---------------------------------------------------------------------------
__global__ __launch_bounds__(256) void combine_kernel(
    const float* __restrict__ h, const float* __restrict__ xc,
    const float* __restrict__ attn, const int* __restrict__ codes,
    float* __restrict__ out) {
  size_t idx = (size_t)blockIdx.x * 256 + threadIdx.x;
  size_t tok = idx / D_;
  int code = codes[tok];
  int c3 = code & 3;
  float hv = h[idx];
  float y;
  if (c3 == 1)
    y = xc[idx];
  else if (c3 == 3)
    y = attn[idx];
  else if (c3 == 0 && (code & 4))
    y = 0.f;
  else
    y = hv;
  out[idx] = y + hv;
}

// ---------------------------------------------------------------------------
extern "C" void kernel_launch(void* const* d_in, const int* in_sizes, int n_in,
                              void* d_out, int out_size, void* d_ws,
                              size_t ws_size, hipStream_t stream) {
  const float* h     = (const float*)d_in[0];
  const float* reg   = (const float*)d_in[1];
  const float* r_w1  = (const float*)d_in[2];
  const float* r_b1  = (const float*)d_in[3];
  const float* r_w2  = (const float*)d_in[4];
  const float* r_b2  = (const float*)d_in[5];
  const float* convw = (const float*)d_in[6];
  const float* convb = (const float*)d_in[7];
  const float* pw_w  = (const float*)d_in[8];
  const float* pw_b  = (const float*)d_in[9];
  const float* cnn_g = (const float*)d_in[10];
  const float* cnn_b = (const float*)d_in[11];
  const float* alpha = (const float*)d_in[12];
  const float* rp_w  = (const float*)d_in[13];
  const float* rp_b  = (const float*)d_in[14];
  const float* reg_g = (const float*)d_in[15];
  const float* reg_b = (const float*)d_in[16];
  const float* q_w   = (const float*)d_in[17];
  const float* q_b   = (const float*)d_in[18];
  const float* k_w   = (const float*)d_in[19];
  const float* k_b   = (const float*)d_in[20];
  const float* v_w   = (const float*)d_in[21];
  const float* v_b   = (const float*)d_in[22];
  const float* o_w   = (const float*)d_in[23];
  const float* o_b   = (const float*)d_in[24];
  const float* rk_w  = (const float*)d_in[25];
  const float* rk_b  = (const float*)d_in[26];
  const float* rv_w  = (const float*)d_in[27];
  const float* rv_b  = (const float*)d_in[28];
  const float* f1_w  = (const float*)d_in[29];
  const float* f1_b  = (const float*)d_in[30];
  const float* f2_w  = (const float*)d_in[31];
  const float* f2_b  = (const float*)d_in[32];
  const float* ln1_g = (const float*)d_in[33];
  const float* ln1_b = (const float*)d_in[34];
  const float* ln2_g = (const float*)d_in[35];
  const float* ln2_b = (const float*)d_in[36];

  float* out = (float*)d_out;
  const size_t FB = (size_t)B_ * S_ * D_;    // 3145728
  const size_t FFB = (size_t)B_ * S_ * FF_;  // 12582912
  const size_t DD = (size_t)D_ * D_;         // 589824
  const size_t DF = (size_t)D_ * FF_;        // 2359296
  const int M = B_ * S_;

  char* w = (char*)d_ws;
  int* codes = (int*)w;       w += sizeof(int) * (size_t)B_ * S_;
  int* counters = (int*)w;    w += 256;
  // f32 scratch
  float* bufX = (float*)w;    w += FB * 4;  // pw out -> xc
  float* bufP = (float*)w;    w += FB * 4;  // rp proj -> o-proj out
  float* bufY = (float*)w;    w += FB * 4;  // ypre/y1
  float* bufG = (float*)w;    w += FB * 4;  // f2 out
  float* regpre = (float*)w;  w += (size_t)B_ * D_ * 4;
  float* Kr = (float*)w;      w += (size_t)B_ * D_ * 4;
  float* Vr = (float*)w;      w += (size_t)B_ * D_ * 4;
  // bf16 activations
  bf16* hb    = (bf16*)w;     w += FB * 2;
  bf16* convo = (bf16*)w;     w += FB * 2;
  bf16* Qb    = (bf16*)w;     w += FB * 2;
  bf16* Kb    = (bf16*)w;     w += FB * 2;
  bf16* Vb    = (bf16*)w;     w += FB * 2;
  bf16* attnb = (bf16*)w;     w += FB * 2;
  bf16* y1b   = (bf16*)w;     w += FB * 2;
  bf16* f1b   = (bf16*)w;     w += FFB * 2;
  // bf16 weights
  bf16* w_pw = (bf16*)w;      w += DD * 2;
  bf16* w_rp = (bf16*)w;      w += DD * 2;
  bf16* w_q  = (bf16*)w;      w += DD * 2;
  bf16* w_k  = (bf16*)w;      w += DD * 2;
  bf16* w_v  = (bf16*)w;      w += DD * 2;
  bf16* w_o  = (bf16*)w;      w += DD * 2;
  bf16* w_f1 = (bf16*)w;      w += DF * 2;
  bf16* w_f2 = (bf16*)w;      w += DF * 2;

  dim3 gD(M / GBM, D_ / GBN);   // 32 x 12
  dim3 gF(M / GBM, FF_ / GBN);  // 32 x 48
  auto cvtg = [](size_t n) { return (int)((n + 255) / 256); };

  init_kernel<<<1, 64, 0, stream>>>(counters);
  // bf16 staging of activations + all GEMM weights
  cvt_kernel<<<cvtg(FB), 256, 0, stream>>>(h, hb, (long)FB);
  cvt_kernel<<<cvtg(DD), 256, 0, stream>>>(pw_w, w_pw, (long)DD);
  cvt_kernel<<<cvtg(DD), 256, 0, stream>>>(rp_w, w_rp, (long)DD);
  cvt_kernel<<<cvtg(DD), 256, 0, stream>>>(q_w, w_q, (long)DD);
  cvt_kernel<<<cvtg(DD), 256, 0, stream>>>(k_w, w_k, (long)DD);
  cvt_kernel<<<cvtg(DD), 256, 0, stream>>>(v_w, w_v, (long)DD);
  cvt_kernel<<<cvtg(DD), 256, 0, stream>>>(o_w, w_o, (long)DD);
  cvt_kernel<<<cvtg(DF), 256, 0, stream>>>(f1_w, w_f1, (long)DF);
  cvt_kernel<<<cvtg(DF), 256, 0, stream>>>(f2_w, w_f2, (long)DF);

  router_kernel<<<B_ * S_, 64, 0, stream>>>(h, r_w1, r_b1, r_w2, r_b2, codes,
                                            counters);
  // local tether CNN path
  dwconv_kernel<<<(int)(FB / 256), 256, 0, stream>>>(h, convw, convb, convo);
  gemm_kernel<<<gD, 256, 0, stream>>>(convo, w_pw, pw_b, bufX, nullptr, M, D_,
                                      D_, 0);
  rowln_kernel<<<M, 256, 0, stream>>>(bufX, nullptr, cnn_g, cnn_b, bufX,
                                      nullptr, D_);
  // register update path
  gemm_kernel<<<gD, 256, 0, stream>>>(hb, w_rp, rp_b, bufP, nullptr, M, D_, D_,
                                      0);
  regupd_kernel<<<dim3(D_ / 256, B_), 256, 0, stream>>>(bufP, codes, counters,
                                                        reg, alpha, regpre);
  rowln_kernel<<<B_, 256, 0, stream>>>(regpre, nullptr, reg_g, reg_b, out + FB,
                                       nullptr, D_);
  // attention path (bf16 Q/K/V for TDM-staged flash attention)
  gemm_kernel<<<gD, 256, 0, stream>>>(hb, w_q, q_b, nullptr, Qb, M, D_, D_, 0);
  gemm_kernel<<<gD, 256, 0, stream>>>(hb, w_k, k_b, nullptr, Kb, M, D_, D_, 0);
  gemm_kernel<<<gD, 256, 0, stream>>>(hb, w_v, v_b, nullptr, Vb, M, D_, D_, 0);
  regkv_kernel<<<(2 * B_ * D_) / 256, 256, 0, stream>>>(reg, rk_w, rk_b, rv_w,
                                                        rv_b, Kr, Vr);
  attn_kernel<<<dim3(S_ / 64, H_, B_), 128, 0, stream>>>(
      Qb, Kb, Vb, k_b, v_b, Kr, Vr, codes, counters, attnb);
  gemm_kernel<<<gD, 256, 0, stream>>>(attnb, w_o, o_b, bufP, nullptr, M, D_,
                                      D_, 0);
  // combine + LN1 (y1 in f32 for residual, bf16 for FFN GEMM)
  combine_kernel<<<(int)(FB / 256), 256, 0, stream>>>(h, bufX, bufP, codes,
                                                      bufY);
  rowln_kernel<<<M, 256, 0, stream>>>(bufY, nullptr, ln1_g, ln1_b, bufY, y1b,
                                      D_);
  // FFN + LN2 -> output
  gemm_kernel<<<gF, 256, 0, stream>>>(y1b, w_f1, f1_b, nullptr, f1b, M, FF_,
                                      D_, 1);
  gemm_kernel<<<gD, 256, 0, stream>>>(f1b, w_f2, f2_b, bufG, nullptr, M, D_,
                                      FF_, 0);
  rowln_kernel<<<M, 256, 0, stream>>>(bufG, bufY, ln2_g, ln2_b, out, nullptr,
                                      D_);
}
